// GateLoopedAttention_12455405158688
// MI455X (gfx1250) — matrix operand order repaired and checked by reading, not compiled, not run on hardware
//
#include <hip/hip_runtime.h>

typedef __attribute__((ext_vector_type(16))) __bf16 v16bf;
typedef __attribute__((ext_vector_type(8)))  __bf16 v8bf;
typedef __attribute__((ext_vector_type(8)))  float  v8f;
typedef __attribute__((ext_vector_type(4)))  int    v4i;

#define DIMC   1024
#define SEQN   2048
#define BATCHN 2
#define NPAD   2052   /* SEQN + KER-1 */
#define OC3    3072
#define KCONV  5120   /* DIMC * KER */
#define HDIM   32
#define BH     64     /* BATCHN * HEADS */
#define NLAYER 32
#define LSTM_D 64     /* ring depth (power of two) */

#define BM 128
#define BN 128
#define BK 64
#define LDT 72        /* padded LDS stride (bf16 elems) */
#define LDG 132       /* padded LDS stride for gate buffer (f32) */

#if defined(__gfx1250__) && __has_builtin(__builtin_amdgcn_global_load_async_to_lds_b128)
#define ASYNC_LDS 1
#else
#define ASYNC_LDS 0
#endif

#define AS1 __attribute__((address_space(1)))
#define AS3 __attribute__((address_space(3)))

__device__ __forceinline__ void wait_async_lds() {
#if ASYNC_LDS
#if __has_builtin(__builtin_amdgcn_s_wait_asynccnt)
  __builtin_amdgcn_s_wait_asynccnt(0);
#else
  asm volatile("s_wait_asynccnt 0x0" ::: "memory");
#endif
#endif
}

// ---------------- fragment loaders (ISA 7.12.2 layouts) ----------------
__device__ __forceinline__ v16bf load_fragA(const __bf16* s, int row, int k0, int lane) {
  // 16-bit A 16x32: lanes 0-15 hold K[0..7],K[16..23]; lanes 16-31 K[8..15],K[24..31]
  int koff = k0 + ((lane >> 4) << 3);
  const __bf16* p = s + row * LDT + koff;
  v8bf lo = *(const v8bf*)p;
  v8bf hi = *(const v8bf*)(p + 16);
  v16bf r;
#pragma unroll
  for (int i = 0; i < 8; ++i) { r[i] = lo[i]; r[i + 8] = hi[i]; }
  return r;
}
__device__ __forceinline__ v16bf load_fragB(const __bf16* s, int col, int k0, int lane) {
  // 16-bit B 32x16: lane n holds column n; lanes 0-15 K[0..7],K[16..23]; lanes 16-31 K[8..15],K[24..31]
  int koff = k0 + ((lane >> 4) << 3);
  const __bf16* p = s + col * LDT + koff;
  v8bf lo = *(const v8bf*)p;
  v8bf hi = *(const v8bf*)(p + 16);
  v16bf r;
#pragma unroll
  for (int i = 0; i < 8; ++i) { r[i] = lo[i]; r[i + 8] = hi[i]; }
  return r;
}

// ---------------- generic bf16 WMMA GEMM: C(MxN,f32) = A(MxK) * B(KxN) + bias ----
// A row addressing: batch = m / mPerBatch, addr = batch*batchStrideA + (m%mPerBatch)*ldA + k
__global__ __launch_bounds__(256) void k_gemm_bf16(
    const __bf16* __restrict__ A, const __bf16* __restrict__ B,
    float* __restrict__ C, const float* __restrict__ bias,
    int M, int N, int K, int mPerBatch, size_t batchStrideA, int ldA, int ldC)
{
  __shared__ __bf16 sA[2][BM * LDT];
  __shared__ __bf16 sB[2][BN * LDT];
  const int tid = threadIdx.x, lane = tid & 31, wv = tid >> 5;
  const int mBase = blockIdx.y * BM, nBase = blockIdx.x * BN;
  const int mW = (wv >> 2) * 64, nW = (wv & 3) * 32;

  v8f acc[4][2];
#pragma unroll
  for (int i = 0; i < 4; ++i)
#pragma unroll
    for (int j = 0; j < 2; ++j)
#pragma unroll
      for (int q = 0; q < 8; ++q) acc[i][j][q] = 0.f;

  const int arow = tid >> 1, acol0 = (tid & 1) * 32;
  const int am = mBase + arow;
  const int abatch = am / mPerBatch, ar = am % mPerBatch;
  const __bf16* aRow = A + (size_t)abatch * batchStrideA + (size_t)ar * ldA;
  const int brow = tid >> 2, bcol0 = (tid & 3) * 32;

#if !ASYNC_LDS
  v8bf areg[4];
#endif
  v8bf breg[4];

  // issue phase: global loads for tile at k0 (A -> async LDS or regs, B -> regs)
  auto issueLoads = [&](int k0, int buf) {
#if ASYNC_LDS
#pragma unroll
    for (int j = 0; j < 4; ++j)
      __builtin_amdgcn_global_load_async_to_lds_b128(
          (AS1 v4i*)(void*)(__bf16*)(aRow + k0 + acol0 + j * 8),
          (AS3 v4i*)(void*)(&sA[buf][arow * LDT + acol0 + j * 8]), 0, 0);
#else
    (void)buf;
#pragma unroll
    for (int j = 0; j < 4; ++j)
      areg[j] = *(const v8bf*)(aRow + k0 + acol0 + j * 8);
#endif
#pragma unroll
    for (int j = 0; j < 4; ++j)
      breg[j] = *(const v8bf*)(B + (size_t)(k0 + brow) * N + nBase + bcol0 + j * 8);
  };
  // commit phase: registers -> LDS (B transposed scatter), then drain async A
  auto commitStores = [&](int buf) {
#if !ASYNC_LDS
#pragma unroll
    for (int j = 0; j < 4; ++j)
      *(v8bf*)(&sA[buf][arow * LDT + acol0 + j * 8]) = areg[j];
#endif
#pragma unroll
    for (int j = 0; j < 4; ++j)
#pragma unroll
      for (int e = 0; e < 8; ++e)
        sB[buf][(bcol0 + j * 8 + e) * LDT + brow] = breg[j][e];
    wait_async_lds();
  };

  issueLoads(0, 0);
  commitStores(0);
  __syncthreads();

  int buf = 0;
  for (int k0 = 0; k0 < K; k0 += BK, buf ^= 1) {
    const int kn = k0 + BK;
    const bool hasNext = kn < K;
    if (hasNext) {
      issueLoads(kn, buf ^ 1);
      if (kn + BK < K) {
        __builtin_prefetch((const void*)(aRow + kn + BK + acol0), 0, 0);
        __builtin_prefetch((const void*)(B + (size_t)(kn + BK + brow) * N + nBase + bcol0), 0, 0);
      }
    }
    const __bf16* cA = sA[buf];
    const __bf16* cB = sB[buf];
#pragma unroll
    for (int ks = 0; ks < 2; ++ks) {
      const int kk = ks * 32;
      v16bf af[4], bfr[2];
#pragma unroll
      for (int mt = 0; mt < 4; ++mt)
        af[mt] = load_fragA(cA, mW + mt * 16 + (lane & 15), kk, lane);
#pragma unroll
      for (int nt = 0; nt < 2; ++nt)
        bfr[nt] = load_fragB(cB, nW + nt * 16 + (lane & 15), kk, lane);
#pragma unroll
      for (int mt = 0; mt < 4; ++mt)
#pragma unroll
        for (int nt = 0; nt < 2; ++nt)
          acc[mt][nt] = __builtin_amdgcn_wmma_f32_16x16x32_bf16(
              false, af[mt], false, bfr[nt], (short)0, acc[mt][nt], false, false);
    }
    if (hasNext) commitStores(buf ^ 1);
    __syncthreads();
  }
  // epilogue: C/D layout — VGPR r: lanes0-15 M=r, lanes16-31 M=8+r; N = lane&15
#pragma unroll
  for (int mt = 0; mt < 4; ++mt) {
    const int rbase = mBase + mW + mt * 16 + ((lane >> 4) << 3);
#pragma unroll
    for (int nt = 0; nt < 2; ++nt) {
      const int col = nBase + nW + nt * 16 + (lane & 15);
      const float bv = bias ? bias[col] : 0.f;
#pragma unroll
      for (int r = 0; r < 8; ++r)
        C[(size_t)(rbase + r) * ldC + col] = acc[mt][nt][r] + bv;
    }
  }
}

// ---------------- prep kernels ----------------
__global__ void k_pad_x(const float* __restrict__ x, __bf16* __restrict__ xpad) {
  size_t i = (size_t)blockIdx.x * 256 + threadIdx.x;
  if (i >= (size_t)BATCHN * NPAD * DIMC) return;
  size_t b = i / ((size_t)NPAD * DIMC);
  size_t r = i % ((size_t)NPAD * DIMC);
  size_t row = r >> 10, c = r & 1023;
  float v = (row < 4) ? 0.f : x[b * ((size_t)SEQN * DIMC) + (row - 4) * DIMC + c];
  xpad[i] = (__bf16)v;
}
__global__ void k_wconv(const float* __restrict__ w, __bf16* __restrict__ wt) {
  size_t i = (size_t)blockIdx.x * 256 + threadIdx.x;
  if (i >= (size_t)KCONV * OC3) return;
  size_t kk = i / OC3, oc = i % OC3;
  size_t tap = kk >> 10, ic = kk & 1023;
  wt[i] = (__bf16)w[oc * (size_t)KCONV + ic * 5 + tap];
}
__global__ void k_wsq(const float* __restrict__ w, __bf16* __restrict__ wt) {
  size_t i = (size_t)blockIdx.x * 256 + threadIdx.x;
  if (i >= (size_t)DIMC * DIMC) return;
  size_t k = i / DIMC, o = i % DIMC;
  wt[i] = (__bf16)w[o * (size_t)DIMC + k];
}
__global__ void k_init(int* prog) { if (threadIdx.x < 64) prog[threadIdx.x] = 0; }

// ---------------- pklatt linear-attention scan: one wave per (b,head) -------
__global__ void k_pklatt(const float* __restrict__ qkv, float* __restrict__ attn) {
  const int bh = blockIdx.x;
  const int b = bh >> 5, h = bh & 31;
  const int e = threadIdx.x; // 0..31, owns column e of kv state
  const float* base = qkv + ((size_t)b * SEQN) * OC3 + h * HDIM + e;
  __shared__ __align__(16) float kq[68]; // interleaved (kf, qf), 33 rows + pad
  float kv[33], kcum[33];
#pragma unroll
  for (int d = 0; d < 33; ++d) { kv[d] = 0.f; kcum[d] = 0.f; }

  for (int t = 0; t < SEQN; ++t) {
    const float* row = base + (size_t)t * OC3;
    float qx = row[0];
    float kx = row[DIMC];
    float vx = row[2 * DIMC];
    float sk = kx * kx, sq = qx * qx;
#pragma unroll
    for (int s = 16; s > 0; s >>= 1) { sk += __shfl_xor(sk, s, 32); sq += __shfl_xor(sq, s, 32); }
    float ak = sqrtf(sk), aq = sqrtf(sq);
    float kf = (1.f - exp2f(-ak)) * kx / fmaxf(ak, 1e-12f);
    float qf = (1.f - exp2f(-aq)) * qx / fmaxf(aq, 1e-12f);
    __syncthreads();
    kq[2 * e] = kf; kq[2 * e + 1] = qf;
    if (e == 0) { kq[64] = 1.f; kq[65] = 1.f; } // ones feature
    __syncthreads();
    float acc = 0.f, qks = 0.f;
#pragma unroll
    for (int d = 0; d < 33; ++d) {
      float2 w = *(const float2*)&kq[2 * d];
      kcum[d] += w.x;
      kv[d]   += w.x * vx;
      qks += w.y * kcum[d];
      acc += w.y * kv[d];
    }
    attn[((size_t)bh * SEQN + t) * HDIM + e] = acc / qks;
  }
}

// ---------------- persistent pipelined LSTM: one WG per layer ----------------
__device__ __forceinline__ float tanh_fast(float x) {
  float a = __expf(-2.f * fabsf(x));
  float t = (1.f - a) / (1.f + a);
  return copysignf(t, x);
}
__global__ __launch_bounds__(256) void k_lstm(
    const float* __restrict__ attn,
    const float* __restrict__ wih, const float* __restrict__ whh,
    const float* __restrict__ bih, const float* __restrict__ bhh,
    float* __restrict__ ring, int* prog, float* __restrict__ lstmo)
{
  const int L = blockIdx.x;
  const int tid = threadIdx.x;
  const int lane = tid & 31, wv = tid >> 5;

  __shared__ __bf16 sW[128 * LDT];   // Wcat[n][k], k<32:wih, k>=32:whh
  __shared__ __bf16 sX[64 * LDT];    // [batch][ x(0..31) | h(32..63) ]
  __shared__ float  sG[64 * LDG];    // gates [batch][128]
  __shared__ float  sBias[128];

  for (int i = tid; i < 128 * 64; i += 256) {
    int n = i >> 6, k = i & 63;
    float w = (k < 32) ? wih[(size_t)L * 4096 + n * 32 + k]
                       : whh[(size_t)L * 4096 + n * 32 + (k - 32)];
    sW[n * LDT + k] = (__bf16)w;
  }
  for (int i = tid; i < 128; i += 256) sBias[i] = bih[L * 128 + i] + bhh[L * 128 + i];
  for (int i = tid; i < 64 * LDT; i += 256) sX[i] = (__bf16)0.f;
  __syncthreads();

  v16bf Bfr[2];
#pragma unroll
  for (int kk = 0; kk < 2; ++kk)
    Bfr[kk] = load_fragB(sW, wv * 16 + (lane & 15), kk * 32, lane);

  float c[8];
#pragma unroll
  for (int i = 0; i < 8; ++i) c[i] = 0.f;
  const int b = tid >> 2;
  const int j0 = (tid & 3) * 8;

  for (int t = 0; t < SEQN; ++t) {
    if (tid == 0) {
      if (L > 0)
        while (__hip_atomic_load(&prog[L - 1], __ATOMIC_ACQUIRE, __HIP_MEMORY_SCOPE_AGENT) < t + 1)
          __builtin_amdgcn_s_sleep(1);
      if (L < NLAYER - 1 && t >= LSTM_D)
        while (__hip_atomic_load(&prog[L + 1], __ATOMIC_ACQUIRE, __HIP_MEMORY_SCOPE_AGENT) < t - LSTM_D + 1)
          __builtin_amdgcn_s_sleep(1);
    }
    __syncthreads();

    const float* xin = (L == 0)
        ? attn + ((size_t)b * SEQN + t) * HDIM + j0
        : ring + ((size_t)(L - 1) * LSTM_D + (t & (LSTM_D - 1))) * (BH * HDIM) + b * HDIM + j0;
    float4 x0 = *(const float4*)xin;
    float4 x1 = *(const float4*)(xin + 4);
    sX[b * LDT + j0 + 0] = (__bf16)x0.x; sX[b * LDT + j0 + 1] = (__bf16)x0.y;
    sX[b * LDT + j0 + 2] = (__bf16)x0.z; sX[b * LDT + j0 + 3] = (__bf16)x0.w;
    sX[b * LDT + j0 + 4] = (__bf16)x1.x; sX[b * LDT + j0 + 5] = (__bf16)x1.y;
    sX[b * LDT + j0 + 6] = (__bf16)x1.z; sX[b * LDT + j0 + 7] = (__bf16)x1.w;
    __syncthreads();

#pragma unroll
    for (int mt = 0; mt < 4; ++mt) {
      v8f acc;
#pragma unroll
      for (int q = 0; q < 8; ++q) acc[q] = 0.f;
#pragma unroll
      for (int kk = 0; kk < 2; ++kk) {
        v16bf af = load_fragA(sX, mt * 16 + (lane & 15), kk * 32, lane);
        acc = __builtin_amdgcn_wmma_f32_16x16x32_bf16(false, af, false, Bfr[kk], (short)0, acc, false, false);
      }
      const int r0 = mt * 16 + ((lane >> 4) << 3);
      const int cc = wv * 16 + (lane & 15);
#pragma unroll
      for (int r = 0; r < 8; ++r) sG[(r0 + r) * LDG + cc] = acc[r];
    }
    __syncthreads();

    float hv[8];
#pragma unroll
    for (int jj = 0; jj < 8; ++jj) {
      const int j = j0 + jj;
      float gi = sG[b * LDG + j]      + sBias[j];
      float gf = sG[b * LDG + 32 + j] + sBias[32 + j];
      float gg = sG[b * LDG + 64 + j] + sBias[64 + j];
      float go = sG[b * LDG + 96 + j] + sBias[96 + j];
      float si = 1.f / (1.f + __expf(-gi));
      float sf = 1.f / (1.f + __expf(-gf));
      float so = 1.f / (1.f + __expf(-go));
      c[jj] = sf * c[jj] + si * tanh_fast(gg);
      hv[jj] = so * tanh_fast(c[jj]);
      sX[b * LDT + 32 + j] = (__bf16)hv[jj];
    }
    if (L == NLAYER - 1) {
      const float* res = attn + ((size_t)b * SEQN + t) * HDIM + j0;
      float4 r0 = *(const float4*)res, r1 = *(const float4*)(res + 4);
      float* op = lstmo + ((size_t)b * SEQN + t) * HDIM + j0;
      *(float4*)op       = make_float4(hv[0] + r0.x, hv[1] + r0.y, hv[2] + r0.z, hv[3] + r0.w);
      *(float4*)(op + 4) = make_float4(hv[4] + r1.x, hv[5] + r1.y, hv[6] + r1.z, hv[7] + r1.w);
    } else {
      float* rp = ring + ((size_t)L * LSTM_D + (t & (LSTM_D - 1))) * (BH * HDIM) + b * HDIM + j0;
      *(float4*)rp       = make_float4(hv[0], hv[1], hv[2], hv[3]);
      *(float4*)(rp + 4) = make_float4(hv[4], hv[5], hv[6], hv[7]);
    }
    __threadfence();
    __syncthreads();
    if (tid == 0)
      __hip_atomic_store(&prog[L], t + 1, __ATOMIC_RELEASE, __HIP_MEMORY_SCOPE_AGENT);
  }
}

// ---------------- silu(x@Wg^T) * merge(lstm_out) -> bf16 A for final GEMM ----
__global__ void k_gate_merge(const float* __restrict__ xw, const float* __restrict__ lo,
                             __bf16* __restrict__ gA) {
  size_t i = (size_t)blockIdx.x * 256 + threadIdx.x;
  if (i >= (size_t)BATCHN * SEQN * DIMC) return;
  size_t m = i >> 10, cidx = i & 1023;
  size_t b = m / SEQN, t = m % SEQN;
  size_t h = cidx >> 5, d = cidx & 31;
  float g = xw[i];
  float s = g / (1.f + __expf(-g));
  float o = lo[(((size_t)(b * 32 + h)) * SEQN + t) * HDIM + d];
  gA[i] = (__bf16)(s * o);
}

// ---------------- launch ----------------
extern "C" void kernel_launch(void* const* d_in, const int* in_sizes, int n_in,
                              void* d_out, int out_size, void* d_ws, size_t ws_size,
                              hipStream_t stream) {
  (void)in_sizes; (void)n_in; (void)out_size; (void)ws_size;
  const float* x      = (const float*)d_in[0];
  const float* w_conv = (const float*)d_in[1];
  const float* b_conv = (const float*)d_in[2];
  const float* w_gate = (const float*)d_in[3];
  const float* w_out  = (const float*)d_in[4];
  const float* w_ih   = (const float*)d_in[5];
  const float* w_hh   = (const float*)d_in[6];
  const float* b_ih   = (const float*)d_in[7];
  const float* b_hh   = (const float*)d_in[8];
  float* out = (float*)d_out;

  char* base = (char*)d_ws;
  size_t off = 0;
  auto alloc = [&](size_t bytes) -> char* {
    char* p = base + off;
    off = (off + bytes + 255) & ~(size_t)255;
    return p;
  };
  __bf16* xpad   = (__bf16*)alloc((size_t)BATCHN * NPAD * DIMC * 2);
  __bf16* wConvT = (__bf16*)alloc((size_t)KCONV * OC3 * 2);
  __bf16* wGateT = (__bf16*)alloc((size_t)DIMC * DIMC * 2);
  __bf16* wOutT  = (__bf16*)alloc((size_t)DIMC * DIMC * 2);
  float*  qkv    = (float*) alloc((size_t)BATCHN * SEQN * OC3 * 4);
  float*  attn   = (float*) alloc((size_t)BH * SEQN * HDIM * 4);
  float*  lstmo  = (float*) alloc((size_t)BH * SEQN * HDIM * 4);
  float*  ring   = (float*) alloc((size_t)(NLAYER - 1) * LSTM_D * BH * HDIM * 4);
  int*    prog   = (int*)   alloc(64 * 4);
  // overlay dead qkv region after pklatt:
  float*  xw = qkv;
  __bf16* gA = (__bf16*)((char*)qkv + (size_t)BATCHN * SEQN * DIMC * 4);

  k_init<<<1, 64, 0, stream>>>(prog);
  {
    size_t tot = (size_t)BATCHN * NPAD * DIMC;
    k_pad_x<<<(unsigned)((tot + 255) / 256), 256, 0, stream>>>(x, xpad);
  }
  {
    size_t tot = (size_t)KCONV * OC3;
    k_wconv<<<(unsigned)((tot + 255) / 256), 256, 0, stream>>>(w_conv, wConvT);
  }
  k_wsq<<<(DIMC * DIMC + 255) / 256, 256, 0, stream>>>(w_gate, wGateT);
  k_wsq<<<(DIMC * DIMC + 255) / 256, 256, 0, stream>>>(w_out, wOutT);

  // causal conv as one linear im2col GEMM: A[m=(b,t)][k] = xpad[b][t*1024+k], K=5120
  k_gemm_bf16<<<dim3(OC3 / BN, (BATCHN * SEQN) / BM), 256, 0, stream>>>(
      xpad, wConvT, qkv, b_conv,
      BATCHN * SEQN, OC3, KCONV, SEQN, (size_t)NPAD * DIMC, DIMC, OC3);

  k_pklatt<<<BH, 32, 0, stream>>>(qkv, attn);

  // gate GEMM (A = un-padded rows of xpad) -> xw (overlaid on qkv)
  k_gemm_bf16<<<dim3(DIMC / BN, (BATCHN * SEQN) / BM), 256, 0, stream>>>(
      xpad + 4 * DIMC, wGateT, xw, nullptr,
      BATCHN * SEQN, DIMC, DIMC, SEQN, (size_t)NPAD * DIMC, DIMC, DIMC);

  k_lstm<<<NLAYER, 256, 0, stream>>>(attn, w_ih, w_hh, b_ih, b_hh, ring, prog, lstmo);

  {
    size_t tot = (size_t)BATCHN * SEQN * DIMC;
    k_gate_merge<<<(unsigned)((tot + 255) / 256), 256, 0, stream>>>(xw, lstmo, gA);
  }

  // final GEMM -> d_out (fp32)
  k_gemm_bf16<<<dim3(DIMC / BN, (BATCHN * SEQN) / BM), 256, 0, stream>>>(
      gA, wOutT, out, nullptr,
      BATCHN * SEQN, DIMC, DIMC, BATCHN * SEQN, 0, DIMC, DIMC);
}
